// MultiHeadAttention_18296560681460
// MI455X (gfx1250) — matrix operand order repaired and checked by reading, not compiled, not run on hardware
//
#include <hip/hip_runtime.h>
#include <hip/hip_bf16.h>

typedef unsigned short u16;
typedef unsigned int   u32;
typedef unsigned long long u64;
typedef __attribute__((ext_vector_type(16))) __bf16 v16bf;
typedef __attribute__((ext_vector_type(8)))  float  v8f;
typedef __attribute__((ext_vector_type(4)))  unsigned int v4u;
typedef __attribute__((ext_vector_type(8)))  int v8i_;
typedef __attribute__((ext_vector_type(4)))  int v4i;

#define SS 1024
#define EE 1024
#define HH 16
#define DD 64

union Frag {
  v16bf v;
  u32   u[8];
};

__device__ inline u16 f2bf(float f) {
  return (u16)((__builtin_bit_cast(u32, f) + 0x8000u) >> 16);
}
// pack two floats -> two bf16 in one dword: round-half-up + v_perm_b32
__device__ inline u32 pk2bf(float a, float b) {
  u32 ua = __builtin_bit_cast(u32, a) + 0x8000u;
  u32 ub = __builtin_bit_cast(u32, b) + 0x8000u;
  return __builtin_amdgcn_perm(ub, ua, 0x07060302u);
}

__device__ inline v8f wmma_bf16(const Frag& a, const Frag& b, v8f c) {
  return __builtin_amdgcn_wmma_f32_16x16x32_bf16(false, a.v, false, b.v,
                                                 (short)0, c, false, false);
}

// A-fragment K offset for bf16 16x32 A layout (pair p in 0..7, lane-half hlf)
__device__ inline int a_koff(int p, int hlf) {
  int j = 2 * p;
  return (j < 8) ? (j + hlf * 8) : (j + 8 + hlf * 8);
}

// gfx1250 async global->LDS copy (ASYNCcnt-tracked path).
__device__ inline void async_b128(u32 lds_off, const void* g) {
  asm volatile("global_load_async_to_lds_b128 %0, %1, off"
               :: "v"(lds_off), "v"(g) : "memory");
}
__device__ inline void wait_async0() {
  asm volatile("s_wait_asynccnt 0x0" ::: "memory");
}
__device__ inline u32 lds_off(const void* p) {
  return (u32)(size_t)p;   // LDS aperture lives in the high 32 bits
}

// ---------------------------------------------------------------------------
// Tensor Data Mover: 2D tile (tile_k x tile_rows) of bf16, global -> LDS,
// with hardware LDS padding: 2 DWORDs pad per 16 DWORDs (64B) => row stride
// 36 u16, matching the GEMM LDS layout. D# built per ISA §8.3/8.4.
// ---------------------------------------------------------------------------
#if __has_builtin(__builtin_amdgcn_tensor_load_to_lds)
#define HAS_TDM 1
__device__ inline void tdm_load_2d(u32 lds_addr, const void* gaddr,
                                   u32 tensor_d0, u32 tensor_d1,
                                   u32 stride_elems, u32 tile_k, u32 tile_rows) {
  u64 ga = (u64)(size_t)gaddr;
  v4u g0;
  g0[0] = 1u;                                 // count=1 (valid user descriptor)
  g0[1] = lds_addr;                           // lds_addr [63:32]
  g0[2] = (u32)ga;                            // global_addr [95:64]
  g0[3] = (u32)(ga >> 32) | (2u << 30);       // global_addr hi + type=2 (image)
  v8i_ g1;
  g1[0] = (int)((1u << 16)       // data_size = 1 (2 bytes)
              | (1u << 20)       // pad_enable
              | (3u << 22)       // pad_interval: 16 DWORDs (64B = tile row)
              | (1u << 25));     // pad_amount: 2 DWORDs (4 u16)
  g1[1] = (int)((tensor_d0 & 0xFFFFu) << 16);                    // dim0 lo
  g1[2] = (int)(((tensor_d0 >> 16) & 0xFFFFu) |
                ((tensor_d1 & 0xFFFFu) << 16));                  // dim0 hi, dim1 lo
  g1[3] = (int)(((tensor_d1 >> 16) & 0xFFFFu) |
                ((tile_k & 0xFFFFu) << 16));                     // dim1 hi, tile_dim0
  g1[4] = (int)(tile_rows & 0xFFFFu);                            // tile_dim1 (dim2=0)
  g1[5] = (int)stride_elems;                                     // dim0_stride lo32
  g1[6] = 0;                                                     // stride hi, dim1_stride
  g1[7] = 0;
  v4i z = {};
#if __clang_major__ >= 23
  v8i_ z8 = {};
  __builtin_amdgcn_tensor_load_to_lds(g0, g1, z, z, z8, 0);
#else
  __builtin_amdgcn_tensor_load_to_lds(g0, g1, z, z, 0);
#endif
}
#else
#define HAS_TDM 0
#endif

__device__ inline void wait_tensor0() {
#if __has_builtin(__builtin_amdgcn_s_wait_tensorcnt)
  __builtin_amdgcn_s_wait_tensorcnt(0);
#else
  asm volatile("s_wait_tensorcnt 0x0" ::: "memory");
#endif
}

// ---------------------------------------------------------------------------
// Kernel 1: convert + transpose all four weight matrices to bf16, [N][K]
// ---------------------------------------------------------------------------
__global__ __launch_bounds__(256) void wtrans_kernel(
    const float* __restrict__ Wq, const float* __restrict__ Wk,
    const float* __restrict__ Wv, const float* __restrict__ Wo,
    u16* __restrict__ wt) {
  int mat = blockIdx.x >> 12;
  int i = ((blockIdx.x & 4095) << 8) + threadIdx.x;   // i = k*1024 + n
  const float* W = (mat == 0) ? Wq : (mat == 1) ? Wk : (mat == 2) ? Wv : Wo;
  int k = i >> 10, n = i & 1023;
  wt[((size_t)mat << 20) + ((size_t)n << 10) + k] = f2bf(W[i]);
}

// ---------------------------------------------------------------------------
// Kernel 2: projection GEMM  out = oscale * X(f32)[M,K] @ Wt(bf16)[N,K]^T
// mode 0: out layout [B,H,S,D]; mode 1: out layout [B,H,D,S]
// Wt tile comes in by TDM; X tile is loaded (4 b128 in flight), converted
// with v_perm_b32 packing, and stored to LDS.
// ---------------------------------------------------------------------------
__global__ __launch_bounds__(256) void gemm_proj_kernel(
    const float* __restrict__ X, const u16* __restrict__ Wt,
    u16* __restrict__ out, int mode, float oscale) {
  __shared__ u16 As[128 * 36];
  __shared__ u16 Bs[128 * 36];
  const int tid = threadIdx.x;
  const int lane = tid & 31, wid = tid >> 5;
  const int hlf = lane >> 4, l16 = lane & 15;
  const int wm = wid & 3, wn = wid >> 2;
  const int mBase = blockIdx.y * 128;
  const int nBase = blockIdx.x * 128;

  v8f acc[2][4] = {};

  for (int kk = 0; kk < EE; kk += 32) {
#if HAS_TDM
    if (wid == 0)                           // one DMA descriptor per block tile
      tdm_load_2d(lds_off(Bs), Wt + (size_t)nBase * EE + kk,
                  EE, EE, EE, 32, 128);
#else
#pragma unroll
    for (int i = 0; i < 2; ++i) {
      int c = tid + i * 256;
      int n = c >> 2, kc = (c & 3) * 8;
      async_b128(lds_off(Bs + n * 36 + kc),
                 Wt + (size_t)(nBase + n) * EE + kk + kc);
    }
#endif
    // stage X tile (fp32 -> bf16), loads first for ILP
    float4 xv[4];
#pragma unroll
    for (int i = 0; i < 4; ++i) {
      int c = tid + i * 256;
      int row = c >> 3, kc = (c & 7) * 4;
      xv[i] = *(const float4*)(X + (size_t)(mBase + row) * EE + kk + kc);
    }
#pragma unroll
    for (int i = 0; i < 4; ++i) {
      int c = tid + i * 256;
      int row = c >> 3, kc = (c & 7) * 4;
      *(u32*)(As + row * 36 + kc)     = pk2bf(xv[i].x, xv[i].y);
      *(u32*)(As + row * 36 + kc + 2) = pk2bf(xv[i].z, xv[i].w);
    }
    if (kk + 32 < EE)
      __builtin_prefetch(X + (size_t)(mBase + (tid >> 3)) * EE + kk + 32, 0, 0);
#if HAS_TDM
    if (wid == 0) wait_tensor0();
#else
    wait_async0();
#endif
    __syncthreads();

    Frag a[2], b[4];
#pragma unroll
    for (int mt = 0; mt < 2; ++mt) {
      int row = wm * 32 + mt * 16 + l16;
#pragma unroll
      for (int p = 0; p < 8; ++p)
        a[mt].u[p] = *(const u32*)(As + row * 36 + a_koff(p, hlf));
    }
#pragma unroll
    for (int nt = 0; nt < 4; ++nt) {
      int col = wn * 64 + nt * 16 + l16;
#pragma unroll
      for (int p = 0; p < 8; ++p)
        b[nt].u[p] = *(const u32*)(Bs + col * 36 + a_koff(p, hlf));
    }
#pragma unroll
    for (int mt = 0; mt < 2; ++mt)
#pragma unroll
      for (int nt = 0; nt < 4; ++nt)
        acc[mt][nt] = wmma_bf16(a[mt], b[nt], acc[mt][nt]);
    __syncthreads();
  }

#pragma unroll
  for (int mt = 0; mt < 2; ++mt)
#pragma unroll
    for (int nt = 0; nt < 4; ++nt)
#pragma unroll
      for (int r = 0; r < 8; ++r) {
        int m = mBase + wm * 32 + mt * 16 + hlf * 8 + r;
        int n = nBase + wn * 64 + nt * 16 + l16;
        int bidx = m >> 10, s = m & 1023;
        int h = n >> 6, d = n & 63;
        size_t idx = (mode == 0)
            ? ((size_t)(bidx * HH + h) * SS + s) * DD + d
            : ((size_t)(bidx * HH + h) * DD + d) * SS + s;
        out[idx] = f2bf(acc[mt][nt][r] * oscale);
      }
}

// ---------------------------------------------------------------------------
// Kernel 3: fused flash attention (unchanged structure; mask tile staged by
// the ASYNCcnt global->LDS path, double-buffered).
// ---------------------------------------------------------------------------
__global__ __launch_bounds__(256) void attn_kernel(
    const u16* __restrict__ qb, const u16* __restrict__ kb,
    const u16* __restrict__ vtb, const unsigned char* __restrict__ mask,
    u16* __restrict__ deg) {
  __shared__ u16 Pl[8][16 * 34];
  __shared__ __align__(16) unsigned char Ml[2][16 * 32];
  const int tid = threadIdx.x;
  const int lane = tid & 31, wid = tid >> 5;
  const int hlf = lane >> 4, l16 = lane & 15;
  const int b = blockIdx.x >> 7;
  const int s16 = (blockIdx.x >> 1) & 63;
  const int hg = blockIdx.x & 1;
  const int h = hg * 8 + wid;
  const int bh = b * HH + h;
  const int sBase = s16 * 16;
  const size_t maskRow = (size_t)b * SS + sBase;

  if (wid == 0) {
    int row = lane >> 1, c16 = (lane & 1) * 16;
    async_b128(lds_off(&Ml[0][row * 32 + c16]),
               mask + (maskRow + row) * SS + c16);
    wait_async0();
  }

  Frag aq[2];
  const u16* qrow = qb + ((size_t)bh * SS + sBase + l16) * DD;
#pragma unroll
  for (int kd = 0; kd < 2; ++kd)
#pragma unroll
    for (int p = 0; p < 8; ++p)
      aq[kd].u[p] = *(const u32*)(qrow + kd * 32 + a_koff(p, hlf));

  v8f o[4] = {};
  float mrow[8], lrow[8];
#pragma unroll
  for (int r = 0; r < 8; ++r) { mrow[r] = -1e30f; lrow[r] = 0.f; }

  __syncthreads();   // mask buffer 0 visible

  for (int tb = 0; tb < SS; tb += 32) {
    const int buf = (tb >> 5) & 1;
    v8f c0 = {}, c1 = {};
#pragma unroll
    for (int th = 0; th < 2; ++th) {
      const u16* krow = kb + ((size_t)bh * SS + tb + th * 16 + l16) * DD;
      Frag bk0, bk1;
#pragma unroll
      for (int p = 0; p < 8; ++p) {
        bk0.u[p] = *(const u32*)(krow + a_koff(p, hlf));
        bk1.u[p] = *(const u32*)(krow + 32 + a_koff(p, hlf));
      }
      v8f cc = {};
      cc = wmma_bf16(aq[1], bk1, cc);
      cc = wmma_bf16(aq[0], bk0, cc);
      if (th == 0) c0 = cc; else c1 = cc;
    }

    float p0[8], p1[8];
#pragma unroll
    for (int r = 0; r < 8; ++r) {
      const unsigned char* mr = &Ml[buf][(hlf * 8 + r) * 32];
      float s0 = c0[r];
      float s1 = c1[r];
      if (mr[l16])      s0 = -1e9f;
      if (mr[16 + l16]) s1 = -1e9f;
      float rm = fmaxf(s0, s1);
      rm = fmaxf(rm, __shfl_xor(rm, 1));
      rm = fmaxf(rm, __shfl_xor(rm, 2));
      rm = fmaxf(rm, __shfl_xor(rm, 4));
      rm = fmaxf(rm, __shfl_xor(rm, 8));
      float mn = fmaxf(mrow[r], rm);
      float al = __expf(mrow[r] - mn);
      float e0 = __expf(s0 - mn);
      float e1 = __expf(s1 - mn);
      float rs = e0 + e1;
      rs += __shfl_xor(rs, 1);
      rs += __shfl_xor(rs, 2);
      rs += __shfl_xor(rs, 4);
      rs += __shfl_xor(rs, 8);
      lrow[r] = lrow[r] * al + rs;
      mrow[r] = mn;
      p0[r] = e0; p1[r] = e1;
#pragma unroll
      for (int dt = 0; dt < 4; ++dt) o[dt][r] *= al;
    }

#pragma unroll
    for (int r = 0; r < 8; ++r) {
      int ml = hlf * 8 + r;
      Pl[wid][ml * 34 + l16]      = f2bf(p0[r]);
      Pl[wid][ml * 34 + 16 + l16] = f2bf(p1[r]);
    }
    __syncthreads();
    Frag pf;
#pragma unroll
    for (int p = 0; p < 8; ++p)
      pf.u[p] = *(const u32*)(&Pl[wid][l16 * 34 + a_koff(p, hlf)]);

    if (wid == 0 && tb + 32 < SS) {
      int row = lane >> 1, c16 = (lane & 1) * 16;
      async_b128(lds_off(&Ml[buf ^ 1][row * 32 + c16]),
                 mask + (maskRow + row) * SS + tb + 32 + c16);
      wait_async0();
    }
    __syncthreads();

#pragma unroll
    for (int dt = 0; dt < 4; ++dt) {
      const u16* vrow = vtb + ((size_t)bh * DD + dt * 16 + l16) * SS + tb;
      Frag bv;
#pragma unroll
      for (int p = 0; p < 8; ++p)
        bv.u[p] = *(const u32*)(vrow + a_koff(p, hlf));
      o[dt] = wmma_bf16(pf, bv, o[dt]);
    }
  }

#pragma unroll
  for (int dt = 0; dt < 4; ++dt)
#pragma unroll
    for (int r = 0; r < 8; ++r) {
      int m = sBase + hlf * 8 + r;
      int d = dt * 16 + l16;
      deg[((size_t)(b * SS + m)) * EE + h * DD + d] = f2bf(o[dt][r] / lrow[r]);
    }
}

// ---------------------------------------------------------------------------
// Kernel 4: output GEMM  tmp = degree(bf16) @ Wo + identity(f32)
// Both tiles arrive by TDM (wave 0 -> A tile, wave 1 -> B tile).
// ---------------------------------------------------------------------------
__global__ __launch_bounds__(256) void gemm_out_kernel(
    const u16* __restrict__ A, const u16* __restrict__ Wt,
    const float* __restrict__ resid, float* __restrict__ out) {
  __shared__ u16 As[128 * 36];
  __shared__ u16 Bs[128 * 36];
  const int tid = threadIdx.x;
  const int lane = tid & 31, wid = tid >> 5;
  const int hlf = lane >> 4, l16 = lane & 15;
  const int wm = wid & 3, wn = wid >> 2;
  const int mBase = blockIdx.y * 128;
  const int nBase = blockIdx.x * 128;

  v8f acc[2][4] = {};

  for (int kk = 0; kk < EE; kk += 32) {
#if HAS_TDM
    if (wid == 0)
      tdm_load_2d(lds_off(As), A + (size_t)mBase * EE + kk,
                  EE, 8192, EE, 32, 128);
    if (wid == 1)
      tdm_load_2d(lds_off(Bs), Wt + (size_t)nBase * EE + kk,
                  EE, EE, EE, 32, 128);
#else
#pragma unroll
    for (int i = 0; i < 2; ++i) {
      int c = tid + i * 256;
      int row = c >> 2, kc = (c & 3) * 8;
      async_b128(lds_off(As + row * 36 + kc),
                 A + (size_t)(mBase + row) * EE + kk + kc);
      async_b128(lds_off(Bs + row * 36 + kc),
                 Wt + (size_t)(nBase + row) * EE + kk + kc);
    }
#endif
    if (kk + 32 < EE)
      __builtin_prefetch(A + (size_t)(mBase + (tid >> 2)) * EE + kk + 32, 0, 0);
#if HAS_TDM
    if (wid < 2) wait_tensor0();
#else
    wait_async0();
#endif
    __syncthreads();

    Frag a[2], b[4];
#pragma unroll
    for (int mt = 0; mt < 2; ++mt) {
      int row = wm * 32 + mt * 16 + l16;
#pragma unroll
      for (int p = 0; p < 8; ++p)
        a[mt].u[p] = *(const u32*)(As + row * 36 + a_koff(p, hlf));
    }
#pragma unroll
    for (int nt = 0; nt < 4; ++nt) {
      int col = wn * 64 + nt * 16 + l16;
#pragma unroll
      for (int p = 0; p < 8; ++p)
        b[nt].u[p] = *(const u32*)(Bs + col * 36 + a_koff(p, hlf));
    }
#pragma unroll
    for (int mt = 0; mt < 2; ++mt)
#pragma unroll
      for (int nt = 0; nt < 4; ++nt)
        acc[mt][nt] = wmma_bf16(a[mt], b[nt], acc[mt][nt]);
    __syncthreads();
  }

#pragma unroll
  for (int mt = 0; mt < 2; ++mt)
#pragma unroll
    for (int nt = 0; nt < 4; ++nt)
#pragma unroll
      for (int r = 0; r < 8; ++r) {
        int m = mBase + wm * 32 + mt * 16 + hlf * 8 + r;
        int n = nBase + wn * 64 + nt * 16 + l16;
        size_t idx = (size_t)m * EE + n;
        out[idx] = acc[mt][nt][r] + resid[idx];
      }
}

// ---------------------------------------------------------------------------
// Kernel 5: LayerNorm over last dim (E = 1024), one block per row
// ---------------------------------------------------------------------------
__global__ __launch_bounds__(256) void ln_kernel(
    const float* __restrict__ x, const float* __restrict__ g,
    const float* __restrict__ be, float* __restrict__ out) {
  __shared__ float red[2][8];
  const int row = blockIdx.x, tid = threadIdx.x;
  const float4 v = *(const float4*)(x + (size_t)row * EE + tid * 4);
  float s = v.x + v.y + v.z + v.w;
  float q = v.x * v.x + v.y * v.y + v.z * v.z + v.w * v.w;
  s += __shfl_xor(s, 1);  q += __shfl_xor(q, 1);
  s += __shfl_xor(s, 2);  q += __shfl_xor(q, 2);
  s += __shfl_xor(s, 4);  q += __shfl_xor(q, 4);
  s += __shfl_xor(s, 8);  q += __shfl_xor(q, 8);
  s += __shfl_xor(s, 16); q += __shfl_xor(q, 16);
  if ((tid & 31) == 0) { red[0][tid >> 5] = s; red[1][tid >> 5] = q; }
  __syncthreads();
  float ts = 0.f, tq = 0.f;
#pragma unroll
  for (int i = 0; i < 8; ++i) { ts += red[0][i]; tq += red[1][i]; }
  float mu  = ts * (1.f / 1024.f);
  float var = tq * (1.f / 1024.f) - mu * mu;
  float inv = rsqrtf(var + 1e-5f);
  const float4 gg = *(const float4*)(g + tid * 4);
  const float4 bb = *(const float4*)(be + tid * 4);
  float4 o;
  o.x = (v.x - mu) * inv * gg.x + bb.x;
  o.y = (v.y - mu) * inv * gg.y + bb.y;
  o.z = (v.z - mu) * inv * gg.z + bb.z;
  o.w = (v.w - mu) * inv * gg.w + bb.w;
  *(float4*)(out + (size_t)row * EE + tid * 4) = o;
}

// ---------------------------------------------------------------------------
extern "C" void kernel_launch(void* const* d_in, const int* in_sizes, int n_in,
                              void* d_out, int out_size, void* d_ws, size_t ws_size,
                              hipStream_t stream) {
  (void)in_sizes; (void)n_in; (void)out_size; (void)ws_size;
  const float* query = (const float*)d_in[0];
  const float* key   = (const float*)d_in[1];
  const float* value = (const float*)d_in[2];
  const unsigned char* maskp = (const unsigned char*)d_in[3];
  const float* Wq = (const float*)d_in[4];
  const float* Wk = (const float*)d_in[5];
  const float* Wv = (const float*)d_in[6];
  const float* Wo = (const float*)d_in[7];
  const float* gamma = (const float*)d_in[8];
  const float* beta  = (const float*)d_in[9];

  unsigned char* ws = (unsigned char*)d_ws;
  u16*   wt   = (u16*)ws;                     //  8 MB : 4 transposed bf16 weights
  u16*   qb   = (u16*)(ws + (8ull  << 20));   // 16 MB : q [B,H,S,D] bf16 (pre-scaled)
  u16*   kbuf = (u16*)(ws + (24ull << 20));   // 16 MB : k [B,H,S,D] bf16
  u16*   vtb  = (u16*)(ws + (40ull << 20));   // 16 MB : v [B,H,D,S] bf16
  u16*   deg  = (u16*)(ws + (56ull << 20));   // 16 MB : attn out [B*S, H*D] bf16
  float* tmp  = (float*)(ws + (72ull << 20)); // 32 MB : pre-LN fp32

  wtrans_kernel<<<16384, 256, 0, stream>>>(Wq, Wk, Wv, Wo, wt);
  dim3 gg(8, 64);
  gemm_proj_kernel<<<gg, 256, 0, stream>>>(query, wt + (0ull << 20), qb, 0, 0.125f);
  gemm_proj_kernel<<<gg, 256, 0, stream>>>(key,   wt + (1ull << 20), kbuf, 0, 1.0f);
  gemm_proj_kernel<<<gg, 256, 0, stream>>>(value, wt + (2ull << 20), vtb, 1, 1.0f);
  attn_kernel<<<1024, 256, 0, stream>>>(qb, kbuf, vtb, maskp, deg);
  gemm_out_kernel<<<gg, 256, 0, stream>>>(deg, wt + (3ull << 20), query, tmp);
  ln_kernel<<<8192, 256, 0, stream>>>(tmp, gamma, beta, (float*)d_out);
}
